// TextDecoderLayer_5033701671242
// MI455X (gfx1250) — compile-verified, matrix operand order checked by reading, not run on hardware
//
#include <hip/hip_runtime.h>
#include <hip/hip_bf16.h>

// ---------------------------------------------------------------------------
// Problem constants (match reference)
// ---------------------------------------------------------------------------
#define BB   4
#define NN   2048
#define MM   576
#define CC   512
#define HH   8
#define KVH  2
#define DD   64
#define EE   8
#define FF   2048
#define TT   (BB * NN)   // 8192 tokens
#define TMV  (BB * MM)   // 2304 vision tokens
#define CAP  8192        // per-expert token capacity

typedef __bf16 bf16;
typedef __attribute__((ext_vector_type(16))) __bf16 v16bf;
typedef __attribute__((ext_vector_type(8)))  float  v8f;

// ---------------------------------------------------------------------------
// Device helpers
// ---------------------------------------------------------------------------
__device__ __forceinline__ v8f zero8() {
  v8f z;
#pragma unroll
  for (int i = 0; i < 8; ++i) z[i] = 0.0f;
  return z;
}

__device__ __forceinline__ v8f wmma_bf16(v16bf a, v16bf b, v8f c) {
  // D = A(16x32 bf16) * B(32x16 bf16) + C(16x16 f32)
  return __builtin_amdgcn_wmma_f32_16x16x32_bf16(
      /*neg_a=*/false, a, /*neg_b=*/false, b,
      /*c_mod=*/(short)0, c, /*reuse_a=*/false, /*reuse_b=*/false);
}

// A-fragment (and row-major B / transposed-V) loader. Tile is 16x32 bf16,
// row-major with given row stride. ISA layout: lane holds row (lane&15);
// packed K pairs: VGPR v<4 -> K = 2v + 8*half ; v>=4 -> K = 16 + 2(v-4) + 8*half.
// Per lane this is two contiguous 16-byte runs -> compiler merges to 2x b128.
__device__ __forceinline__ v16bf load_frag(const bf16* __restrict__ base,
                                           int stride, int lane) {
  const int r = lane & 15;
  const int half = lane >> 4;
  const bf16* row = base + (size_t)r * stride;
  v16bf f;
#pragma unroll
  for (int v = 0; v < 8; ++v) {
    const int k0 = (v < 4) ? (2 * v + 8 * half) : (16 + 2 * (v - 4) + 8 * half);
    f[2 * v]     = row[k0];
    f[2 * v + 1] = row[k0 + 1];
  }
  return f;
}

__device__ __forceinline__ float gelu_tanh(float x) {
  const float x3 = x * x * x;
  return 0.5f * x * (1.0f + tanhf(0.7978845608028654f * (x + 0.044715f * x3)));
}

// ---------------------------------------------------------------------------
// Elementwise / setup kernels
// ---------------------------------------------------------------------------
__global__ void cast_transpose_kernel(const float* __restrict__ W,
                                      bf16* __restrict__ Wt, int K, int Nc) {
  size_t i = (size_t)blockIdx.x * blockDim.x + threadIdx.x;
  size_t total = (size_t)K * Nc;
  if (i >= total) return;
  int k = (int)(i / Nc);
  int n = (int)(i % Nc);
  Wt[(size_t)n * K + k] = (bf16)W[i];
}

__global__ void cast_bf16_kernel(const float* __restrict__ a,
                                 bf16* __restrict__ o, size_t n) {
  size_t i = (size_t)blockIdx.x * blockDim.x + threadIdx.x;
  if (i < n) o[i] = (bf16)a[i];
}

__global__ void copyf_kernel(const float* __restrict__ a,
                             float* __restrict__ o, size_t n) {
  size_t i = (size_t)blockIdx.x * blockDim.x + threadIdx.x;
  if (i < n) o[i] = a[i];
}

__global__ void zero_router_state_kernel(int* counts, float* psum) {
  int i = threadIdx.x;
  if (i < EE) { counts[i] = 0; psum[i] = 0.0f; }
}

// ---------------------------------------------------------------------------
// LayerNorm (fp32 in -> bf16 out)
// ---------------------------------------------------------------------------
__global__ void ln_bf16_kernel(const float* __restrict__ x,
                               const float* __restrict__ g,
                               const float* __restrict__ b,
                               bf16* __restrict__ out) {
  const int t = blockIdx.x;
  const float* row = x + (size_t)t * CC;
  __shared__ float red[32];
  float s = 0.0f, s2 = 0.0f;
  for (int c = threadIdx.x; c < CC; c += blockDim.x) {
    float v = row[c];
    s += v; s2 += v * v;
  }
#pragma unroll
  for (int off = 16; off; off >>= 1) {
    s  += __shfl_xor(s,  off, 32);
    s2 += __shfl_xor(s2, off, 32);
  }
  const int wave = threadIdx.x >> 5, lane = threadIdx.x & 31;
  if (lane == 0) { red[wave] = s; red[8 + wave] = s2; }
  __syncthreads();
  if (threadIdx.x == 0) {
    float a = 0.0f, c2 = 0.0f;
    for (int i = 0; i < 8; ++i) { a += red[i]; c2 += red[8 + i]; }
    red[16] = a; red[17] = c2;
  }
  __syncthreads();
  const float mean = red[16] * (1.0f / CC);
  const float var  = red[17] * (1.0f / CC) - mean * mean;
  const float inv  = rsqrtf(var + 1e-6f);
  for (int c = threadIdx.x; c < CC; c += blockDim.x) {
    float v = (row[c] - mean) * inv * g[c] + b[c];
    out[(size_t)t * CC + c] = (bf16)v;
  }
}

// ---------------------------------------------------------------------------
// Tiled bf16 GEMM: out[M][Nc] = A[M][K] @ Wt[Nc][K]^T + bias (+ residual)
// block = 256 (8 waves), each wave one 16x16 tile; macro tile 128 x 16.
// transLD != 0: bf16 output stored transposed, out[col][row], ld = transLD
// (used to produce V^T so flash attention's V fragments are contiguous).
// ---------------------------------------------------------------------------
__global__ void gemm_bf16_kernel(const bf16* __restrict__ A,
                                 const bf16* __restrict__ Wt,
                                 const float* __restrict__ bias,
                                 const float* __restrict__ residual,
                                 float* __restrict__ outF,
                                 bf16* __restrict__ outB,
                                 int Mrows, int K, int Nc, int transLD) {
  const int lane = threadIdx.x & 31;
  const int wave = threadIdx.x >> 5;
  const int m0 = blockIdx.x * 128 + wave * 16;
  const int n0 = blockIdx.y * 16;
  if (m0 >= Mrows) return;

  const bf16* Ab = A  + (size_t)m0 * K;
  const bf16* Wb = Wt + (size_t)n0 * K;

  v8f acc = zero8();
  for (int k = 0; k < K; k += 32) {
    if (k + 64 < K) {
      __builtin_prefetch(Ab + k + 64, 0, 1);
      __builtin_prefetch(Wb + k + 64, 0, 1);
    }
    v16bf a = load_frag(Ab + k, K, lane);
    v16bf b = load_frag(Wb + k, K, lane);
    acc = wmma_bf16(a, b, acc);
  }

  const int col  = n0 + (lane & 15);
  const int half = lane >> 4;
  const float bv = bias ? bias[col] : 0.0f;
#pragma unroll
  for (int j = 0; j < 8; ++j) {
    const int row = m0 + j + 8 * half;
    float v = acc[j] + bv;
    if (residual) v += residual[(size_t)row * Nc + col];
    if (outF)         outF[(size_t)row * Nc + col] = v;
    else if (transLD) outB[(size_t)col * transLD + row] = (bf16)v;
    else              outB[(size_t)row * Nc + col] = (bf16)v;
  }
}

// ---------------------------------------------------------------------------
// Flash attention (GQA). One wave per (batch, head, 16-query tile).
// q: [B*n_q][512] (head h at cols h*64..)
// k: [B*n_ctx][128] (kv head g at cols g*64..)
// vt: transposed V, [128][B*n_ctx] with leading dim vld = B*n_ctx
// out: bf16 [B*n_q][512]
// ---------------------------------------------------------------------------
__global__ void flash_attn_kernel(const bf16* __restrict__ qb,
                                  const bf16* __restrict__ kb,
                                  const bf16* __restrict__ vt,
                                  bf16* __restrict__ ob,
                                  int n_q, int n_ctx, int vld, int causal) {
  const int lane = threadIdx.x & 31;
  const int wid  = (blockIdx.x * blockDim.x + threadIdx.x) >> 5;
  const int qtiles = n_q / 16;
  const int qt = wid % qtiles;
  const int h  = (wid / qtiles) % HH;
  const int b  = wid / (qtiles * HH);
  const int g  = h >> 2;                       // H/KVH = 4
  const int half = lane >> 4;
  const int c16  = lane & 15;

  __shared__ bf16 ldsP[4][16 * 32];            // per-wave P staging
  bf16* P = ldsP[threadIdx.x >> 5];

  const bf16* qbase = qb + ((size_t)(b * n_q + qt * 16)) * CC + h * DD;
  const v16bf aq0 = load_frag(qbase,      CC, lane);   // K = d 0..31
  const v16bf aq1 = load_frag(qbase + 32, CC, lane);   // K = d 32..63

  const bf16* kall = kb + (size_t)(b * n_ctx) * (KVH * DD) + g * DD;
  const bf16* vall = vt + (size_t)(g * DD) * vld + (size_t)b * n_ctx;

  v8f accO[4];
#pragma unroll
  for (int i = 0; i < 4; ++i) accO[i] = zero8();
  float mrow[8], lrow[8];
#pragma unroll
  for (int j = 0; j < 8; ++j) { mrow[j] = -1e30f; lrow[j] = 0.0f; }

  const int t_end = causal ? ((qt * 16 + 15) / 32 + 1) : (n_ctx / 32);

  for (int t = 0; t < t_end; ++t) {
    const bf16* kb0 = kall + (size_t)(t * 32) * (KVH * DD);
    v8f S0 = zero8(), S1 = zero8();
    {
      v16bf bk;
      bk = load_frag(kb0,                           KVH * DD, lane); S0 = wmma_bf16(aq0, bk, S0);
      bk = load_frag(kb0 + 32,                      KVH * DD, lane); S0 = wmma_bf16(aq1, bk, S0);
      bk = load_frag(kb0 + 16 * (KVH * DD),         KVH * DD, lane); S1 = wmma_bf16(aq0, bk, S1);
      bk = load_frag(kb0 + 16 * (KVH * DD) + 32,    KVH * DD, lane); S1 = wmma_bf16(aq1, bk, S1);
    }
#pragma unroll
    for (int j = 0; j < 8; ++j) {
      const int row  = qt * 16 + j + 8 * half;
      const int col0 = t * 32 + c16;
      const int col1 = col0 + 16;
      float s0 = S0[j] * 0.125f;                 // D^-0.5
      float s1 = S1[j] * 0.125f;
      if (causal) {
        if (col0 > row) s0 = -1e9f;
        if (col1 > row) s1 = -1e9f;
      }
      float mx = fmaxf(s0, s1);
#pragma unroll
      for (int off = 1; off < 16; off <<= 1) mx = fmaxf(mx, __shfl_xor(mx, off, 32));
      const float mn   = fmaxf(mrow[j], mx);
      const float corr = __expf(mrow[j] - mn);
      const float p0   = __expf(s0 - mn);
      const float p1   = __expf(s1 - mn);
      float rs = p0 + p1;
#pragma unroll
      for (int off = 1; off < 16; off <<= 1) rs += __shfl_xor(rs, off, 32);
      lrow[j] = lrow[j] * corr + rs;
      mrow[j] = mn;
#pragma unroll
      for (int nt = 0; nt < 4; ++nt) accO[nt][j] *= corr;
      P[(j + 8 * half) * 32 + c16]      = (bf16)p0;
      P[(j + 8 * half) * 32 + 16 + c16] = (bf16)p1;
    }
    asm volatile("s_wait_dscnt 0" ::: "memory");   // P writes visible to frag reload
    const v16bf ap = load_frag(P, 32, lane);
    // V^T fragment: rows = d-columns (lane&15), K pairs contiguous (kv tokens)
    const bf16* vb0 = vall + t * 32;
#pragma unroll
    for (int nt = 0; nt < 4; ++nt) {
      v16bf bv = load_frag(vb0 + (size_t)(nt * 16) * vld, vld, lane);
      accO[nt] = wmma_bf16(ap, bv, accO[nt]);
    }
  }

  bf16* obase = ob + ((size_t)(b * n_q + qt * 16)) * CC + h * DD;
#pragma unroll
  for (int nt = 0; nt < 4; ++nt) {
#pragma unroll
    for (int j = 0; j < 8; ++j) {
      const float o = accO[nt][j] / lrow[j];
      obase[(size_t)(j + 8 * half) * CC + nt * 16 + c16] = (bf16)o;
    }
  }
}

// ---------------------------------------------------------------------------
// Router: softmax over E=8, top-2, renormalized gates, dispatch lists + aux sums
// one wave per token (8 tokens per 256-thread block)
// ---------------------------------------------------------------------------
__global__ void router_kernel(const bf16* __restrict__ hln,
                              const float* __restrict__ rw,
                              const float* __restrict__ rb,
                              int* __restrict__ counts,
                              int* __restrict__ lists,
                              float* __restrict__ gates,
                              float* __restrict__ psum) {
  const int wave = threadIdx.x >> 5;
  const int lane = threadIdx.x & 31;
  const int tok  = blockIdx.x * 8 + wave;
  if (tok >= TT) return;

  const bf16* row = hln + (size_t)tok * CC;
  float lg[EE] = {0, 0, 0, 0, 0, 0, 0, 0};
  for (int c = lane; c < CC; c += 32) {
    const float x = (float)row[c];
    const float* w = rw + (size_t)c * EE;
#pragma unroll
    for (int e = 0; e < EE; ++e) lg[e] += x * w[e];
  }
#pragma unroll
  for (int e = 0; e < EE; ++e)
#pragma unroll
    for (int off = 16; off; off >>= 1) lg[e] += __shfl_xor(lg[e], off, 32);

  if (lane == 0) {
    float mx = -1e30f;
#pragma unroll
    for (int e = 0; e < EE; ++e) { lg[e] += rb[e]; mx = fmaxf(mx, lg[e]); }
    float ex[EE], s = 0.0f;
#pragma unroll
    for (int e = 0; e < EE; ++e) { ex[e] = __expf(lg[e] - mx); s += ex[e]; }
    float p[EE];
    const float invs = 1.0f / s;
#pragma unroll
    for (int e = 0; e < EE; ++e) p[e] = ex[e] * invs;

    int i0 = 0;
#pragma unroll
    for (int e = 1; e < EE; ++e) if (p[e] > p[i0]) i0 = e;
    int i1 = (i0 == 0) ? 1 : 0;
#pragma unroll
    for (int e = 0; e < EE; ++e) if (e != i0 && p[e] > p[i1]) i1 = e;

    float g0 = p[i0], g1 = p[i1];
    const float gs = 1.0f / (g0 + g1);
    g0 *= gs; g1 *= gs;

    int pos0 = atomicAdd(&counts[i0], 1);
    lists[i0 * CAP + pos0] = tok;
    gates[i0 * CAP + pos0] = g0;
    int pos1 = atomicAdd(&counts[i1], 1);
    lists[i1 * CAP + pos1] = tok;
    gates[i1 * CAP + pos1] = g1;
#pragma unroll
    for (int e = 0; e < EE; ++e) atomicAdd(&psum[e], p[e]);
  }
}

// ---------------------------------------------------------------------------
// MoE expert MLP, fused: X(16xC) -> GELU(X@W1+b1) -> (@W2+b2) -> gated scatter.
// grid (512 tiles, 8 experts), block 256 (8 waves).
// h is computed in two 1024-wide phases to keep LDS = 16KB(X)+32KB(H) < 64KB.
// ---------------------------------------------------------------------------
__global__ void moe_expert_kernel(const bf16* __restrict__ hln,
                                  const bf16* __restrict__ w1t,   // [E][F][C]
                                  const bf16* __restrict__ w2t,   // [E][C][F]
                                  const float* __restrict__ b1,   // [E][F]
                                  const float* __restrict__ b2,   // [E][C]
                                  const int* __restrict__ counts,
                                  const int* __restrict__ lists,
                                  const float* __restrict__ gates,
                                  float* __restrict__ out) {
  const int e    = blockIdx.y;
  const int tile = blockIdx.x;
  const int cnt  = counts[e];
  if (tile * 16 >= cnt) return;
  int rows = cnt - tile * 16;
  if (rows > 16) rows = 16;

  __shared__ bf16  X[16 * CC];          // 16 KB
  __shared__ bf16  Hs[16 * 1024];       // 32 KB
  __shared__ int   toks[16];
  __shared__ float gts[16];

  const int tid = threadIdx.x;
  if (tid < 16) {
    const int idx = e * CAP + tile * 16 + tid;
    toks[tid] = (tid < rows) ? lists[idx] : 0;
    gts[tid]  = (tid < rows) ? gates[idx] : 0.0f;
  }
  __syncthreads();
  for (int i = tid; i < 16 * CC; i += 256) {
    const int r = i >> 9, c = i & (CC - 1);
    X[i] = (r < rows) ? hln[(size_t)toks[r] * CC + c] : (bf16)0.0f;
  }
  __syncthreads();

  const int lane = tid & 31, wave = tid >> 5;
  const int half = lane >> 4, c16 = lane & 15;

  v8f accOut[4];
#pragma unroll
  for (int i = 0; i < 4; ++i) accOut[i] = zero8();

  const bf16* W1e = w1t + (size_t)e * FF * CC;
  const bf16* W2e = w2t + (size_t)e * CC * FF;

  for (int phase = 0; phase < 2; ++phase) {
    // h tile: this phase covers F cols [phase*1024, phase*1024+1024)
    for (int s = 0; s < 8; ++s) {
      const int nt = phase * 64 + wave * 8 + s;       // 16-col tile index in F
      v8f acc = zero8();
      const bf16* Wb = W1e + (size_t)(nt * 16) * CC;
      for (int k = 0; k < CC; k += 32) {
        if (k + 64 < CC) __builtin_prefetch(Wb + k + 64, 0, 1);
        v16bf a = load_frag(X + k, CC, lane);
        v16bf b = load_frag(Wb + k, CC, lane);
        acc = wmma_bf16(a, b, acc);
      }
      const int col = nt * 16 + c16;
#pragma unroll
      for (int j = 0; j < 8; ++j) {
        float v = acc[j] + b1[e * FF + col];
        v = gelu_tanh(v);
        Hs[(size_t)(j + 8 * half) * 1024 + (nt - phase * 64) * 16 + c16] = (bf16)v;
      }
    }
    __syncthreads();
    // out accumulation: K slice [phase*1024, +1024)
    for (int s2 = 0; s2 < 4; ++s2) {
      const int nt2 = wave * 4 + s2;                  // 16-col tile index in C
      const bf16* Wb2 = W2e + (size_t)(nt2 * 16) * FF + phase * 1024;
      for (int k = 0; k < 1024; k += 32) {
        if (k + 64 < 1024) __builtin_prefetch(Wb2 + k + 64, 0, 1);
        v16bf a = load_frag(Hs + k, 1024, lane);
        v16bf b = load_frag(Wb2 + k, FF, lane);
        accOut[s2] = wmma_bf16(a, b, accOut[s2]);
      }
    }
    __syncthreads();
  }

#pragma unroll
  for (int s2 = 0; s2 < 4; ++s2) {
    const int nt2 = wave * 4 + s2;
    const int col = nt2 * 16 + c16;
#pragma unroll
    for (int j = 0; j < 8; ++j) {
      const int r = j + 8 * half;
      if (r < rows) {
        const float v = accOut[s2][j] + b2[e * CC + col];
        atomicAdd(&out[(size_t)toks[r] * CC + col], gts[r] * v);
      }
    }
  }
}

// ---------------------------------------------------------------------------
// Aux loss: E * sum_e (count_e / T) * (psum_e / T)
// ---------------------------------------------------------------------------
__global__ void finalize_aux_kernel(const int* __restrict__ counts,
                                    const float* __restrict__ psum,
                                    float* __restrict__ out_aux) {
  if (threadIdx.x == 0 && blockIdx.x == 0) {
    const float invT = 1.0f / (float)TT;
    float a = 0.0f;
    for (int e = 0; e < EE; ++e) a += ((float)counts[e] * invT) * (psum[e] * invT);
    out_aux[0] = (float)EE * a;
  }
}

// ---------------------------------------------------------------------------
// Host launch
// ---------------------------------------------------------------------------
extern "C" void kernel_launch(void* const* d_in, const int* in_sizes, int n_in,
                              void* d_out, int out_size, void* d_ws, size_t ws_size,
                              hipStream_t stream) {
  (void)in_sizes; (void)n_in; (void)out_size; (void)ws_size;

  const float* x        = (const float*)d_in[0];
  const float* vis      = (const float*)d_in[1];
  const float* ln1_g    = (const float*)d_in[2];
  const float* ln1_b    = (const float*)d_in[3];
  const float* ln2_g    = (const float*)d_in[4];
  const float* ln2_b    = (const float*)d_in[5];
  const float* ln3_g    = (const float*)d_in[6];
  const float* ln3_b    = (const float*)d_in[7];
  const float* sa_wq = (const float*)d_in[8];  const float* sa_bq = (const float*)d_in[9];
  const float* sa_wk = (const float*)d_in[10]; const float* sa_bk = (const float*)d_in[11];
  const float* sa_wv = (const float*)d_in[12]; const float* sa_bv = (const float*)d_in[13];
  const float* sa_wo = (const float*)d_in[14]; const float* sa_bo = (const float*)d_in[15];
  const float* ca_wq = (const float*)d_in[16]; const float* ca_bq = (const float*)d_in[17];
  const float* ca_wk = (const float*)d_in[18]; const float* ca_bk = (const float*)d_in[19];
  const float* ca_wv = (const float*)d_in[20]; const float* ca_bv = (const float*)d_in[21];
  const float* ca_wo = (const float*)d_in[22]; const float* ca_bo = (const float*)d_in[23];
  const float* router_w = (const float*)d_in[24];
  const float* router_b = (const float*)d_in[25];
  const float* e_w1 = (const float*)d_in[26];  const float* e_b1 = (const float*)d_in[27];
  const float* e_w2 = (const float*)d_in[28];  const float* e_b2 = (const float*)d_in[29];

  float* outF = (float*)d_out;

  // workspace bump allocator
  char* ws = (char*)d_ws;
  size_t off = 0;
  auto alloc = [&](size_t bytes) -> void* {
    void* p = ws + off;
    off = (off + bytes + 255) & ~(size_t)255;
    return p;
  };

  bf16* wt_sa_q = (bf16*)alloc((size_t)CC * CC * 2);
  bf16* wt_sa_k = (bf16*)alloc((size_t)(KVH * DD) * CC * 2);
  bf16* wt_sa_v = (bf16*)alloc((size_t)(KVH * DD) * CC * 2);
  bf16* wt_sa_o = (bf16*)alloc((size_t)CC * CC * 2);
  bf16* wt_ca_q = (bf16*)alloc((size_t)CC * CC * 2);
  bf16* wt_ca_k = (bf16*)alloc((size_t)(KVH * DD) * CC * 2);
  bf16* wt_ca_v = (bf16*)alloc((size_t)(KVH * DD) * CC * 2);
  bf16* wt_ca_o = (bf16*)alloc((size_t)CC * CC * 2);
  bf16* wt_e1   = (bf16*)alloc((size_t)EE * FF * CC * 2);
  bf16* wt_e2   = (bf16*)alloc((size_t)EE * CC * FF * 2);
  bf16* vis_bf  = (bf16*)alloc((size_t)TMV * CC * 2);
  bf16* hln     = (bf16*)alloc((size_t)TT * CC * 2);
  bf16* q_bf    = (bf16*)alloc((size_t)TT * CC * 2);
  bf16* k_bf    = (bf16*)alloc((size_t)TT * (KVH * DD) * 2);
  bf16* vt_bf   = (bf16*)alloc((size_t)TT * (KVH * DD) * 2);   // V^T, [128][rows]
  bf16* attn_bf = (bf16*)alloc((size_t)TT * CC * 2);
  float* x1     = (float*)alloc((size_t)TT * CC * 4);
  float* x2     = (float*)alloc((size_t)TT * CC * 4);
  int*   counts = (int*)alloc(EE * 4);
  int*   lists  = (int*)alloc((size_t)EE * CAP * 4);
  float* gatesb = (float*)alloc((size_t)EE * CAP * 4);
  float* psum   = (float*)alloc(EE * 4);

  const int THR = 256;
  auto lt = [&](const float* W, bf16* Wt, int K, int Nc) {
    size_t tot = (size_t)K * Nc;
    cast_transpose_kernel<<<(unsigned)((tot + THR - 1) / THR), THR, 0, stream>>>(W, Wt, K, Nc);
  };

  // --- weight conversion (fp32 -> bf16, transposed to [N][K]) ---
  lt(sa_wq, wt_sa_q, CC, CC);
  lt(sa_wk, wt_sa_k, CC, KVH * DD);
  lt(sa_wv, wt_sa_v, CC, KVH * DD);
  lt(sa_wo, wt_sa_o, CC, CC);
  lt(ca_wq, wt_ca_q, CC, CC);
  lt(ca_wk, wt_ca_k, CC, KVH * DD);
  lt(ca_wv, wt_ca_v, CC, KVH * DD);
  lt(ca_wo, wt_ca_o, CC, CC);
  for (int e = 0; e < EE; ++e) {
    lt(e_w1 + (size_t)e * CC * FF, wt_e1 + (size_t)e * FF * CC, CC, FF);
    lt(e_w2 + (size_t)e * FF * CC, wt_e2 + (size_t)e * CC * FF, FF, CC);
  }
  {
    size_t tot = (size_t)TMV * CC;
    cast_bf16_kernel<<<(unsigned)((tot + THR - 1) / THR), THR, 0, stream>>>(vis, vis_bf, tot);
  }

  // --- 1) causal self-attention ---
  ln_bf16_kernel<<<TT, THR, 0, stream>>>(x, ln1_g, ln1_b, hln);
  {
    dim3 gq(TT / 128, CC / 16);
    gemm_bf16_kernel<<<gq, THR, 0, stream>>>(hln, wt_sa_q, sa_bq, nullptr, nullptr, q_bf, TT, CC, CC, 0);
    dim3 gkv(TT / 128, (KVH * DD) / 16);
    gemm_bf16_kernel<<<gkv, THR, 0, stream>>>(hln, wt_sa_k, sa_bk, nullptr, nullptr, k_bf, TT, CC, KVH * DD, 0);
    gemm_bf16_kernel<<<gkv, THR, 0, stream>>>(hln, wt_sa_v, sa_bv, nullptr, nullptr, vt_bf, TT, CC, KVH * DD, TT);
  }
  {
    const int waves = BB * HH * (NN / 16);   // 4096
    flash_attn_kernel<<<waves / 4, 128, 0, stream>>>(q_bf, k_bf, vt_bf, attn_bf, NN, NN, TT, 1);
  }
  {
    dim3 go(TT / 128, CC / 16);
    gemm_bf16_kernel<<<go, THR, 0, stream>>>(attn_bf, wt_sa_o, sa_bo, x, x1, nullptr, TT, CC, CC, 0);
  }

  // --- 2) cross-attention ---
  ln_bf16_kernel<<<TT, THR, 0, stream>>>(x1, ln2_g, ln2_b, hln);
  {
    dim3 gq(TT / 128, CC / 16);
    gemm_bf16_kernel<<<gq, THR, 0, stream>>>(hln, wt_ca_q, ca_bq, nullptr, nullptr, q_bf, TT, CC, CC, 0);
    dim3 gkv(TMV / 128, (KVH * DD) / 16);
    gemm_bf16_kernel<<<gkv, THR, 0, stream>>>(vis_bf, wt_ca_k, ca_bk, nullptr, nullptr, k_bf, TMV, CC, KVH * DD, 0);
    gemm_bf16_kernel<<<gkv, THR, 0, stream>>>(vis_bf, wt_ca_v, ca_bv, nullptr, nullptr, vt_bf, TMV, CC, KVH * DD, TMV);
  }
  {
    const int waves = BB * HH * (NN / 16);
    flash_attn_kernel<<<waves / 4, 128, 0, stream>>>(q_bf, k_bf, vt_bf, attn_bf, NN, MM, TMV, 0);
  }
  {
    dim3 go(TT / 128, CC / 16);
    gemm_bf16_kernel<<<go, THR, 0, stream>>>(attn_bf, wt_ca_o, ca_bo, x1, x2, nullptr, TT, CC, CC, 0);
  }

  // --- 3) MoE ---
  ln_bf16_kernel<<<TT, THR, 0, stream>>>(x2, ln3_g, ln3_b, hln);
  zero_router_state_kernel<<<1, 32, 0, stream>>>(counts, psum);
  {
    size_t tot = (size_t)TT * CC;
    copyf_kernel<<<(unsigned)((tot + THR - 1) / THR), THR, 0, stream>>>(x2, outF, tot);
  }
  router_kernel<<<TT / 8, THR, 0, stream>>>(hln, router_w, router_b, counts, lists, gatesb, psum);
  {
    dim3 gm(CAP / 16, EE);
    moe_expert_kernel<<<gm, THR, 0, stream>>>(hln, wt_e1, wt_e2, e_b1, e_b2,
                                              counts, lists, gatesb, outF);
  }
  finalize_aux_kernel<<<1, 32, 0, stream>>>(counts, psum, outF + (size_t)TT * CC);
}
